// VRNN_17523466567927
// MI455X (gfx1250) — matrix operand order repaired and checked by reading, not compiled, not run on hardware
//
#include <hip/hip_runtime.h>

typedef __attribute__((ext_vector_type(16))) __bf16 v16bf;
typedef __attribute__((ext_vector_type(8)))  __bf16 v8bf;
typedef __attribute__((ext_vector_type(8)))  float  v8f;

#define B_ 128
#define T_ 512
#define H_ 256
#define Z_ 64
#define O_ 200
#define X_ 200

// ---------------- WMMA helpers ----------------

__device__ inline v8f wmma_bf16(v16bf a, v16bf b, v8f c){
  return __builtin_amdgcn_wmma_f32_16x16x32_bf16(false, a, false, b, (short)0, c, false, false);
}

// A operand (16xK, row-major bf16, LDS or global). ISA 16-bit A layout:
// lane L (row = L&15) holds K = kbase..kbase+7 and kbase+16..kbase+23, kbase=(L>>4)*8.
__device__ inline v16bf load_a16(const __bf16* A, int ld, int kb, int lane){
  const int row = lane & 15;
  const int k0  = kb*32 + ((lane>>4)<<3);
  const __bf16* p = A + (size_t)row*ld + k0;
  v8bf lo = *(const v8bf*)p;
  v8bf hi = *(const v8bf*)(p + 16);
  return __builtin_shufflevector(lo, hi, 0,1,2,3,4,5,6,7,8,9,10,11,12,13,14,15);
}

// B operand from transposed weights WT[N][Kpad] (bf16): lane L -> column n = nt*16+(L&15),
// K = kbase..kbase+7 and kbase+16..kbase+23, kbase = kb*32 + (L>>4)*8 (matches A layout).
__device__ inline v16bf load_b16(const __bf16* WT, int kpad, int kb, int ntile, int lane){
  const int n  = ntile*16 + (lane & 15);
  const int k0 = kb*32 + ((lane>>4)<<3);
  const __bf16* p = WT + (size_t)n*kpad + k0;
  v8bf lo = *(const v8bf*)p;
  v8bf hi = *(const v8bf*)(p + 16);
  return __builtin_shufflevector(lo, hi, 0,1,2,3,4,5,6,7,8,9,10,11,12,13,14,15);
}

__device__ inline v8f gemm_tile(const __bf16* A, int a_ld, const __bf16* WT, int kpad,
                                int K, int ntile, int lane){
  v8f acc = {};
  for (int kb = 0; kb < (K>>5); ++kb){
    v16bf a = load_a16(A, a_ld, kb, lane);
    v16bf b = load_b16(WT, kpad, kb, ntile, lane);
    acc = wmma_bf16(a, b, acc);
  }
  return acc;
}

// C/D layout: VGPR i -> row (i + (lane>=16 ? 8 : 0)), col nt*16 + (lane&15)
__device__ inline void store_relu_bf16(v8f acc, float bias, __bf16* out, int ld, int ntile, int lane){
  const int col = ntile*16 + (lane & 15);
  const int rb  = (lane>>4)<<3;
#pragma unroll
  for (int i=0;i<8;++i){
    float v = acc[i] + bias;
    out[(size_t)(rb+i)*ld + col] = (__bf16)(v > 0.f ? v : 0.f);
  }
}

__device__ inline float sigmoidf_(float x){ return 1.f/(1.f + expf(-x)); }
// softplus(log2 * a)/log2 + 1e-3  ==  log2(1 + 2^a) + 1e-3
__device__ inline float softplus2_(float a){
  float v = (a > 20.f) ? a : log2f(1.f + exp2f(a));
  return v + 1e-3f;
}

// ---------------- weight conversion: fp32 [K][Nsrc] -> bf16 transposed [Ndst][Kpad] ----------------

__global__ void convert_wt(const float* __restrict__ src, __bf16* __restrict__ dst,
                           int K, int Nsrc, int Ndst, int Kpad){
  int idx = blockIdx.x*blockDim.x + threadIdx.x;
  int total = Ndst*Kpad;
  if (idx >= total) return;
  int n = idx / Kpad, k = idx % Kpad;
  float v = (k < K && n < Nsrc) ? src[(size_t)k*Nsrc + n] : 0.f;
  dst[idx] = (__bf16)v;
}

// ---------------- phi_x (time-parallel) ----------------

// layer 0: A = x fp32 [M][200], converted on the fly; weights Kpad=224 zero padded
__device__ inline v16bf load_a_f32(const float* A, int ldA, int K, int kb, int lane){
  const int row = lane & 15;
  const int k0  = kb*32 + ((lane>>4)<<3);
  v16bf out = {};
#pragma unroll
  for (int j=0;j<8;++j){
    int ka = k0 + j;
    int kc = k0 + 16 + j;
    out[j]   = (ka < K) ? (__bf16)A[(size_t)row*ldA + ka] : (__bf16)0.f;
    out[j+8] = (kc < K) ? (__bf16)A[(size_t)row*ldA + kc] : (__bf16)0.f;
  }
  return out;
}

__global__ __launch_bounds__(256) void phix_l0(const float* __restrict__ x,
                                               const __bf16* __restrict__ WT,
                                               const float* __restrict__ bias,
                                               __bf16* __restrict__ out){
  const int wave = threadIdx.x >> 5, lane = threadIdx.x & 31;
  const int tile = blockIdx.x*8 + wave;           // 65536 tiles
  const int mt = tile >> 4, nt = tile & 15;
  const float* Ar = x + (size_t)mt*16*X_;
  v8f acc = {};
  for (int kb = 0; kb < 7; ++kb){                 // Kpad 224
    v16bf a = load_a_f32(Ar, X_, X_, kb, lane);
    v16bf b = load_b16(WT, 224, kb, nt, lane);
    acc = wmma_bf16(a, b, acc);
  }
  store_relu_bf16(acc, bias[nt*16 + (lane&15)], out + (size_t)mt*16*H_, H_, nt, lane);
}

__global__ __launch_bounds__(256) void phix_layer(const __bf16* __restrict__ A,
                                                  const __bf16* __restrict__ WT,
                                                  const float* __restrict__ bias,
                                                  __bf16* __restrict__ out){
  const int wave = threadIdx.x >> 5, lane = threadIdx.x & 31;
  const int tile = blockIdx.x*8 + wave;
  const int mt = tile >> 4, nt = tile & 15;
  v8f acc = gemm_tile(A + (size_t)mt*16*H_, H_, WT, H_, H_, nt, lane);
  store_relu_bf16(acc, bias[nt*16 + (lane&15)], out + (size_t)mt*16*H_, H_, nt, lane);
}

// ---------------- persistent recurrent kernel ----------------

struct RecParams {
  const __bf16* phiX;           // [B*T][256] bf16
  const float*  eps;            // [B*T*64]
  // weights (bf16, transposed N x Kpad)
  const __bf16 *pr0,*pr1,*pr2,*prm,*prs;
  const __bf16 *en0,*en1,*en2,*enm,*ens;
  const __bf16 *pz0,*pz1,*pz2,*pz3;
  const __bf16 *de0,*de1,*de2,*dlw;
  const __bf16 *wih,*whh;
  // biases (fp32, from inputs)
  const float *pr0b,*pr1b,*pr2b,*prmb,*prsb;
  const float *en0b,*en1b,*en2b,*enmb,*ensb;
  const float *pz0b,*pz1b,*pz2b,*pz3b;
  const float *de0b,*de1b,*de2b,*dlb;
  const float *bih,*bhh;
  // outputs
  float *oz,*odl,*oem,*oes,*opm,*ops,*okld;
};

__device__ inline void layer16(const __bf16* A, int a_ld, const __bf16* WT, int kpad, int K,
                               const float* bias, __bf16* out, int out_ld, int wave, int lane){
  v8f acc = gemm_tile(A, a_ld, WT, kpad, K, wave, lane);
  store_relu_bf16(acc, bias[wave*16 + (lane&15)], out, out_ld, wave, lane);
}

__device__ inline void head64(const __bf16* A, const __bf16* WT, const float* bias,
                              float* ldsOut, float* gOut, int b0, int t, int nt, int lane, bool sp){
  v8f acc = gemm_tile(A, 512, WT, 256, 256, nt, lane);
  const int col = nt*16 + (lane&15);
  const int rb  = (lane>>4)<<3;
  const float bv = bias[col];
#pragma unroll
  for (int i=0;i<8;++i){
    float v = acc[i] + bv;
    if (sp) v = softplus2_(v);
    ldsOut[(rb+i)*Z_ + col] = v;
    gOut[((size_t)(b0+rb+i)*T_ + t)*Z_ + col] = v;
  }
}

__global__ __launch_bounds__(512,1) void vrnn_recurrent(RecParams P){
  extern __shared__ __align__(32) char smem[];
  float*  h32 = (float*)smem;                       // 16x256 f32
  float*  emL = h32 + 16*H_;                        // 16x64 each
  float*  esL = emL + 16*Z_;
  float*  pmL = esL + 16*Z_;
  float*  psL = pmL + 16*Z_;
  __bf16* hbf = (__bf16*)(psL + 16*Z_);             // 16x256
  __bf16* px  = hbf + 16*H_;                        // 16x256
  __bf16* pzb = px  + 16*H_;                        // 16x256
  __bf16* a0  = pzb + 16*H_;                        // 16x512
  __bf16* a1  = a0  + 16*512;                       // 16x512

  const int tid  = threadIdx.x;
  const int wave = tid >> 5, lane = tid & 31;
  const int b0   = blockIdx.x * 16;

  for (int i = tid; i < 16*H_; i += 512){ h32[i] = 0.f; hbf[i] = (__bf16)0.f; }
  __syncthreads();

  for (int t = 0; t < T_; ++t){
    // phi_x_t tile -> LDS
    for (int i = tid; i < 16*H_; i += 512){
      int r = i >> 8, c = i & 255;
      px[i] = P.phiX[((size_t)(b0+r)*T_ + t)*H_ + c];
    }
    __syncthreads();

    // ---- prior MLP: hbf -> a1 -> a0 -> a1 ----
    layer16(hbf, H_, P.pr0, H_, H_, P.pr0b, a1, 512, wave, lane);  __syncthreads();
    layer16(a1, 512, P.pr1, H_, H_, P.pr1b, a0, 512, wave, lane);  __syncthreads();
    layer16(a0, 512, P.pr2, H_, H_, P.pr2b, a1, 512, wave, lane);  __syncthreads();

    // ---- prior heads (waves 0..7) || stage concat(px,h) into a0 (waves 8..15) ----
    if (wave < 4){
      head64(a1, P.prm, P.prmb, pmL, P.opm, b0, t, wave, lane, false);
    } else if (wave < 8){
      head64(a1, P.prs, P.prsb, psL, P.ops, b0, t, wave-4, lane, true);
    } else {
      for (int i = tid - 256; i < 16*H_; i += 256){
        int r = i >> 8, c = i & 255;
        a0[r*512 + c]       = px[i];
        a0[r*512 + 256 + c] = hbf[i];
      }
    }
    __syncthreads();

    // ---- enc MLP: a0(K=512) -> a1 -> a0 -> a1 ----
    layer16(a0, 512, P.en0, 512, 512, P.en0b, a1, 512, wave, lane); __syncthreads();
    layer16(a1, 512, P.en1, H_, H_, P.en1b, a0, 512, wave, lane);   __syncthreads();
    layer16(a0, 512, P.en2, H_, H_, P.en2b, a1, 512, wave, lane);   __syncthreads();

    // ---- enc heads ----
    if (wave < 4)      head64(a1, P.enm, P.enmb, emL, P.oem, b0, t, wave,   lane, false);
    else if (wave < 8) head64(a1, P.ens, P.ensb, esL, P.oes, b0, t, wave-4, lane, true);
    __syncthreads();

    // ---- z = em + es*eps : write global + stage bf16 into a0[:, :64] ----
    for (int i = tid; i < 16*Z_; i += 512){
      int r = i >> 6, c = i & 63;
      float e  = P.eps[((size_t)(b0+r)*T_ + t)*Z_ + c];
      float zv = emL[i] + esL[i]*e;
      P.oz[((size_t)(b0+r)*T_ + t)*Z_ + c] = zv;
      a0[r*512 + c] = (__bf16)zv;
    }
    __syncthreads();

    // ---- phi_z MLP: a0(K=64) -> a1 -> a0 -> a1 -> pzb ----
    layer16(a0, 512, P.pz0, Z_, Z_, P.pz0b, a1, 512, wave, lane);  __syncthreads();
    layer16(a1, 512, P.pz1, H_, H_, P.pz1b, a0, 512, wave, lane);  __syncthreads();
    layer16(a0, 512, P.pz2, H_, H_, P.pz2b, a1, 512, wave, lane);  __syncthreads();
    layer16(a1, 512, P.pz3, H_, H_, P.pz3b, pzb, H_, wave, lane);  __syncthreads();

    // ---- dec MLP: concat(pz,h) -> a0; a0 -> a1 -> a0 -> a1 ----
    for (int i = tid; i < 16*H_; i += 512){
      int r = i >> 8, c = i & 255;
      a0[r*512 + c]       = pzb[i];
      a0[r*512 + 256 + c] = hbf[i];
    }
    __syncthreads();
    layer16(a0, 512, P.de0, 512, 512, P.de0b, a1, 512, wave, lane); __syncthreads();
    layer16(a1, 512, P.de1, H_, H_, P.de1b, a0, 512, wave, lane);   __syncthreads();
    layer16(a0, 512, P.de2, H_, H_, P.de2b, a1, 512, wave, lane);   __syncthreads();

    // ---- dec_logits (waves 0..12, Npad=208) || stage GRU xin=concat(px,pz) (waves 13..15) ----
    if (wave < 13){
      v8f acc = gemm_tile(a1, 512, P.dlw, H_, H_, wave, lane);
      const int col = wave*16 + (lane&15);
      const int rb  = (lane>>4)<<3;
      if (col < O_){
        const float bv = P.dlb[col];
#pragma unroll
        for (int i=0;i<8;++i)
          P.odl[((size_t)(b0+rb+i)*T_ + t)*O_ + col] = acc[i] + bv;
      }
    } else {
      for (int i = tid - 13*32; i < 16*H_; i += 96){
        int r = i >> 8, c = i & 255;
        a0[r*512 + c]       = px[i];
        a0[r*512 + 256 + c] = pzb[i];
      }
    }
    __syncthreads();

    // ---- GRU: wave w owns gate column tile w; all register-resident ----
    {
      v8f ir = gemm_tile(a0, 512, P.wih, 512, 512, wave,      lane);
      v8f iz = gemm_tile(a0, 512, P.wih, 512, 512, wave + 16, lane);
      v8f in_= gemm_tile(a0, 512, P.wih, 512, 512, wave + 32, lane);
      v8f hr = gemm_tile(hbf, H_, P.whh, H_, H_,   wave,      lane);
      v8f hz = gemm_tile(hbf, H_, P.whh, H_, H_,   wave + 16, lane);
      v8f hn = gemm_tile(hbf, H_, P.whh, H_, H_,   wave + 32, lane);
      const int col = wave*16 + (lane&15);
      const int rb  = (lane>>4)<<3;
      const float bir = P.bih[col], biz = P.bih[H_+col], bin = P.bih[2*H_+col];
      const float bhr = P.bhh[col], bhz = P.bhh[H_+col], bhn = P.bhh[2*H_+col];
      float hold[8];
#pragma unroll
      for (int i=0;i<8;++i) hold[i] = h32[(rb+i)*H_ + col];
      __syncthreads();   // all reads of hbf/h32 complete before update
#pragma unroll
      for (int i=0;i<8;++i){
        float r  = sigmoidf_(ir[i] + bir + hr[i] + bhr);
        float zg = sigmoidf_(iz[i] + biz + hz[i] + bhz);
        float nn = tanhf(in_[i] + bin + r*(hn[i] + bhn));
        float hv = (1.f - zg)*nn + zg*hold[i];
        h32[(rb+i)*H_ + col] = hv;
        hbf[(rb+i)*H_ + col] = (__bf16)hv;
      }
    }
    __syncthreads();

    // ---- KLD ----
    for (int i = tid; i < 16*Z_; i += 512){
      int r = i >> 6, c = i & 63;
      float psv = psL[i], esv = esL[i], emv = emL[i], pmv = pmL[i];
      float d = emv - pmv;
      float kv = logf(psv) - logf(esv) + (esv*esv + d*d)/(2.f*psv*psv) - 0.5f;
      P.okld[((size_t)(b0+r)*T_ + t)*Z_ + c] = kv;
    }
    __syncthreads();
  }
}

// ---------------- host ----------------

extern "C" void kernel_launch(void* const* d_in, const int* in_sizes, int n_in,
                              void* d_out, int out_size, void* d_ws, size_t ws_size,
                              hipStream_t stream) {
  (void)in_sizes; (void)n_in; (void)out_size; (void)ws_size;
  const float* x   = (const float*)d_in[0];
  const float* eps = (const float*)d_in[1];
  auto F = [&](int i){ return (const float*)d_in[i]; };

  __bf16* wsb = (__bf16*)d_ws;
  size_t off = 0;
  auto alloc = [&](size_t n){ __bf16* p = wsb + off; off += n; return p; };
  auto conv = [&](const float* src, __bf16* dst, int K, int Nsrc, int Ndst, int Kpad){
    int total = Ndst*Kpad;
    convert_wt<<<dim3((total+255)/256), dim3(256), 0, stream>>>(src, dst, K, Nsrc, Ndst, Kpad);
  };

  // ---- bf16 transposed weights in workspace ----
  __bf16* pxw0 = alloc(256*224); __bf16* pxw1 = alloc(256*256);
  __bf16* pxw2 = alloc(256*256); __bf16* pxw3 = alloc(256*256);
  __bf16* pzw0 = alloc(256*64);  __bf16* pzw1 = alloc(256*256);
  __bf16* pzw2 = alloc(256*256); __bf16* pzw3 = alloc(256*256);
  __bf16* enw0 = alloc(256*512); __bf16* enw1 = alloc(256*256); __bf16* enw2 = alloc(256*256);
  __bf16* enm  = alloc(64*256);  __bf16* ens  = alloc(64*256);
  __bf16* prw0 = alloc(256*256); __bf16* prw1 = alloc(256*256); __bf16* prw2 = alloc(256*256);
  __bf16* prm  = alloc(64*256);  __bf16* prs  = alloc(64*256);
  __bf16* dew0 = alloc(256*512); __bf16* dew1 = alloc(256*256); __bf16* dew2 = alloc(256*256);
  __bf16* dlw  = alloc(208*256);
  __bf16* wih  = alloc(768*512); __bf16* whh  = alloc(768*256);
  __bf16* phiA = alloc((size_t)B_*T_*H_);
  __bf16* phiB = alloc((size_t)B_*T_*H_);

  conv(F(2),  pxw0, 200, 256, 256, 224);
  conv(F(4),  pxw1, 256, 256, 256, 256);
  conv(F(6),  pxw2, 256, 256, 256, 256);
  conv(F(8),  pxw3, 256, 256, 256, 256);
  conv(F(10), pzw0,  64, 256, 256,  64);
  conv(F(12), pzw1, 256, 256, 256, 256);
  conv(F(14), pzw2, 256, 256, 256, 256);
  conv(F(16), pzw3, 256, 256, 256, 256);
  conv(F(18), enw0, 512, 256, 256, 512);
  conv(F(20), enw1, 256, 256, 256, 256);
  conv(F(22), enw2, 256, 256, 256, 256);
  conv(F(24), enm,  256,  64,  64, 256);
  conv(F(26), ens,  256,  64,  64, 256);
  conv(F(28), prw0, 256, 256, 256, 256);
  conv(F(30), prw1, 256, 256, 256, 256);
  conv(F(32), prw2, 256, 256, 256, 256);
  conv(F(34), prm,  256,  64,  64, 256);
  conv(F(36), prs,  256,  64,  64, 256);
  conv(F(38), dew0, 512, 256, 256, 512);
  conv(F(40), dew1, 256, 256, 256, 256);
  conv(F(42), dew2, 256, 256, 256, 256);
  conv(F(44), dlw,  256, 200, 208, 256);
  conv(F(46), wih,  512, 768, 768, 512);
  conv(F(47), whh,  256, 768, 768, 256);

  // ---- phi_x: 4 layers, 65536 rows, one 16x16 tile per wave ----
  phix_l0   <<<dim3(8192), dim3(256), 0, stream>>>(x,    pxw0, F(3), phiA);
  phix_layer<<<dim3(8192), dim3(256), 0, stream>>>(phiA, pxw1, F(5), phiB);
  phix_layer<<<dim3(8192), dim3(256), 0, stream>>>(phiB, pxw2, F(7), phiA);
  phix_layer<<<dim3(8192), dim3(256), 0, stream>>>(phiA, pxw3, F(9), phiB);

  // ---- recurrent persistent kernel: 8 independent batch blocks ----
  float* out = (float*)d_out;
  const size_t nz  = (size_t)B_*T_*Z_;
  const size_t ndl = (size_t)B_*T_*O_;

  RecParams P;
  P.phiX = phiB; P.eps = eps;
  P.pr0 = prw0; P.pr1 = prw1; P.pr2 = prw2; P.prm = prm; P.prs = prs;
  P.en0 = enw0; P.en1 = enw1; P.en2 = enw2; P.enm = enm; P.ens = ens;
  P.pz0 = pzw0; P.pz1 = pzw1; P.pz2 = pzw2; P.pz3 = pzw3;
  P.de0 = dew0; P.de1 = dew1; P.de2 = dew2; P.dlw = dlw;
  P.wih = wih;  P.whh = whh;
  P.pr0b = F(29); P.pr1b = F(31); P.pr2b = F(33); P.prmb = F(35); P.prsb = F(37);
  P.en0b = F(19); P.en1b = F(21); P.en2b = F(23); P.enmb = F(25); P.ensb = F(27);
  P.pz0b = F(11); P.pz1b = F(13); P.pz2b = F(15); P.pz3b = F(17);
  P.de0b = F(39); P.de1b = F(41); P.de2b = F(43); P.dlb = F(45);
  P.bih = F(48);  P.bhh = F(49);
  P.oz  = out;
  P.odl = out + nz;
  P.oem = out + nz + ndl;
  P.oes = out + 2*nz + ndl;
  P.opm = out + 3*nz + ndl;
  P.ops = out + 4*nz + ndl;
  P.okld= out + 5*nz + ndl;

  const int ldsBytes = 16*H_*4 + 4*16*Z_*4 + 3*16*H_*2 + 2*16*512*2; // 90112
  vrnn_recurrent<<<dim3(8), dim3(512), ldsBytes, stream>>>(P);
}
